// MultiheadedPairedAttention_62088047231063
// MI455X (gfx1250) — matrix operand order repaired and checked by reading, not compiled, not run on hardware
//
#include <hip/hip_runtime.h>
#include <hip/hip_bf16.h>

typedef __attribute__((ext_vector_type(16))) _Float16 v16h;
typedef __attribute__((ext_vector_type(8)))  float    v8f;

#define SDIM 8      // S == T
#define CKD  32
#define CQD  32
#define AD   16
#define ND   32
#define SP   4096   // 16*16*16 spatial
#define NC   1024   // N * CQ

// A-matrix (16x32 f16) element->K mapping per CDNA5 ISA 7.12.2:
// lanes 0-15: V0..3 = K0..7, V4..7 = K16..23 ; lanes 16-31: K8..15, K24..31
__device__ __forceinline__ int a_kmap(int e, int lane) {
  int kofs = (lane & 16) ? 8 : 0;
  int p = e >> 1, hi = e & 1;
  return (p < 4) ? (kofs + 2 * p + hi) : (16 + kofs + 2 * (p - 4) + hi);
}

// CDNA5 async global->LDS copy (ASYNCcnt-tracked, bypasses VGPRs).
// vdst = LDS byte offset (low 32 bits of flat LDS pointer), vaddr = 64-bit global addr.
__device__ __forceinline__ void async_copy_b128(unsigned lds_off, const void* gaddr) {
  asm volatile("global_load_async_to_lds_b128 %0, %1, off"
               :: "v"(lds_off), "v"(gaddr) : "memory");
}
__device__ __forceinline__ void wait_async() {
  asm volatile("s_wait_asynccnt 0" ::: "memory");
}

// ---------------------------------------------------------------------------
// Kernel 0: one-time f32 -> f16 weight conversion
// ---------------------------------------------------------------------------
__global__ void __launch_bounds__(256)
cvt_kernel(const float* __restrict__ src, _Float16* __restrict__ dst, int n) {
  int i = blockIdx.x * 256 + threadIdx.x;
  if (i < n) dst[i] = (_Float16)src[i];
}

// ---------------------------------------------------------------------------
// Kernel 1: GroupNorm(1 group) for key (blocks 0-7) and query (blocks 8-15).
// Emits f16, pre-transposed to [yz][s*16+x][c] so each (y,z) tile of 8KB is
// contiguous for the attention kernel's async LDS fill.
// ---------------------------------------------------------------------------
__global__ void __launch_bounds__(256)
gn_kernel(const float* __restrict__ key, const float* __restrict__ query,
          const float* __restrict__ gkw, const float* __restrict__ gkb,
          const float* __restrict__ gqw, const float* __restrict__ gqb,
          _Float16* __restrict__ nk16, _Float16* __restrict__ nq16) {
  __shared__ float s_sum[256], s_sq[256];
  int gid = blockIdx.x;
  bool isq = gid >= SDIM;
  int s = isq ? gid - SDIM : gid;
  const float* src = (isq ? query : key) + (size_t)s * CKD * SP;
  const float* w  = isq ? gqw : gkw;
  const float* bb = isq ? gqb : gkb;
  _Float16* dst = isq ? nq16 : nk16;
  int tid = threadIdx.x;
  float sum = 0.f, sq = 0.f;
  for (int i = tid; i < CKD * SP; i += 256) { float v = src[i]; sum += v; sq += v * v; }
  s_sum[tid] = sum; s_sq[tid] = sq;
  __syncthreads();
  for (int st = 128; st > 0; st >>= 1) {
    if (tid < st) { s_sum[tid] += s_sum[tid + st]; s_sq[tid] += s_sq[tid + st]; }
    __syncthreads();
  }
  float m  = s_sum[0] * (1.f / (CKD * SP));
  float vr = s_sq[0] * (1.f / (CKD * SP)) - m * m;
  float rs = rsqrtf(vr + 1e-5f);
  for (int i = tid; i < CKD * SP; i += 256) {
    int c  = i >> 12;                   // i = c*4096 + x*256 + yz
    int x  = (i >> 8) & 15;
    int yz = i & 255;
    float v = (src[i] - m) * rs * w[c] + bb[c];
    dst[((size_t)yz * 128 + s * 16 + x) * CKD + c] = (_Float16)v;
  }
}

// ---------------------------------------------------------------------------
// Kernel 2: fused QKV projection (WMMA) + att + softmax(over X) + feat + relu
// one block per (head n, y, z); 128 threads = 4 waves; X=16 voxels local.
// Input tiles pulled with async global->LDS copies (contiguous 8KB each).
// ---------------------------------------------------------------------------
__global__ void __launch_bounds__(128)
attn_kernel(const _Float16* __restrict__ nk16, const _Float16* __restrict__ nq16,
            const _Float16* __restrict__ wk16, const float* __restrict__ bk,
            const _Float16* __restrict__ wq16, const float* __restrict__ bq,
            const _Float16* __restrict__ wv16, const float* __restrict__ bv,
            _Float16* __restrict__ sa) {
  __shared__ _Float16 s_nk[128 * 32];   // row = s*16+x, col = c
  __shared__ _Float16 s_nq[128 * 32];
  __shared__ float    s_k[128 * 16];
  __shared__ float    s_q[128 * 16];
  __shared__ float    s_v[128 * 32];
  __shared__ float    s_att[SDIM * SDIM * 16];   // [(s*8+t)*16 + x]

  int bid = blockIdx.x;
  int n  = bid >> 8;
  int yz = bid & 255;                    // y*16+z
  int tid  = threadIdx.x;
  int lane = tid & 31;
  int wid  = tid >> 5;

  // async copy of the two contiguous 8KB tiles into LDS (512 x 16B chunks each)
  const _Float16* gk = nk16 + (size_t)yz * (128 * CKD);
  const _Float16* gq = nq16 + (size_t)yz * (128 * CKD);
#pragma unroll
  for (int it = 0; it < 4; ++it) {
    int chunk = it * 128 + tid;          // 0..511, 8 halves (16B) per chunk
    async_copy_b128((unsigned)(size_t)(const void*)&s_nk[chunk * 8], gk + chunk * 8);
    async_copy_b128((unsigned)(size_t)(const void*)&s_nq[chunk * 8], gq + chunk * 8);
  }
  wait_async();
  __syncthreads();

  // 32 WMMA tiles: 8 K (task 0-7), 8 Q (8-15), 16 V (16-31); wave-uniform split
  int m     = lane & 15;
  int nn    = lane & 15;
  int kbase = (lane & 16) ? 16 : 0;
  int moff  = (lane & 16) ? 8 : 0;
  for (int task = wid; task < 32; task += 4) {
    union { v16h v; _Float16 h[16]; } afr, bfr;
    union { v8f v; float f[8]; } acc = {};
    if (task < 8) {                         // K projection, s-tile = task
      int rbase = task * 16;
#pragma unroll
      for (int e = 0; e < 16; ++e) {
        afr.h[e] = s_nk[(rbase + m) * 32 + a_kmap(e, lane)];
        bfr.h[e] = wk16[(n * AD + nn) * CKD + a_kmap(e, lane)];
      }
      acc.v = __builtin_amdgcn_wmma_f32_16x16x32_f16(false, afr.v, false, bfr.v,
                                                     (short)0, acc.v, false, false);
      float bias = bk[n * AD + nn];
#pragma unroll
      for (int r = 0; r < 8; ++r) s_k[(rbase + moff + r) * 16 + nn] = acc.f[r] + bias;
    } else if (task < 16) {                 // Q projection, t-tile = task-8
      int rbase = (task - 8) * 16;
#pragma unroll
      for (int e = 0; e < 16; ++e) {
        afr.h[e] = s_nq[(rbase + m) * 32 + a_kmap(e, lane)];
        bfr.h[e] = wq16[(n * AD + nn) * CKD + a_kmap(e, lane)];
      }
      acc.v = __builtin_amdgcn_wmma_f32_16x16x32_f16(false, afr.v, false, bfr.v,
                                                     (short)0, acc.v, false, false);
      float bias = bq[n * AD + nn];
#pragma unroll
      for (int r = 0; r < 8; ++r) s_q[(rbase + moff + r) * 16 + nn] = acc.f[r] + bias;
    } else {                                // V projection: row tile + 2 col tiles
      int tv = task - 16;
      int rbase = (tv >> 1) * 16;
      int o0 = (tv & 1) * 16;
#pragma unroll
      for (int e = 0; e < 16; ++e) {
        afr.h[e] = s_nk[(rbase + m) * 32 + a_kmap(e, lane)];
        bfr.h[e] = wv16[(n * CQD + o0 + nn) * CKD + a_kmap(e, lane)];
      }
      acc.v = __builtin_amdgcn_wmma_f32_16x16x32_f16(false, afr.v, false, bfr.v,
                                                     (short)0, acc.v, false, false);
      float bias = bv[n * CQD + o0 + nn];
#pragma unroll
      for (int r = 0; r < 8; ++r) s_v[(rbase + moff + r) * 32 + o0 + nn] = acc.f[r] + bias;
    }
  }
  __syncthreads();

  // att[s,t,x] = (k . q)/sqrt(A)   thread = (x, t)
  int x = tid & 15, t = tid >> 4;
  for (int s = 0; s < SDIM; ++s) {
    float sum = 0.f;
    for (int a = 0; a < AD; ++a)
      sum += s_k[(s * 16 + x) * 16 + a] * s_q[(t * 16 + x) * 16 + a];
    s_att[(s * SDIM + t) * 16 + x] = sum * 0.25f;
  }
  __syncthreads();

  // softmax over X (16 values) for each (s,t)
  if (tid < 64) {
    float* row = &s_att[tid * 16];
    float mx = row[0];
    for (int i = 1; i < 16; ++i) mx = fmaxf(mx, row[i]);
    float e[16], sum = 0.f;
    for (int i = 0; i < 16; ++i) { e[i] = __expf(row[i] - mx); sum += e[i]; }
    float inv = 1.f / sum;
    for (int i = 0; i < 16; ++i) row[i] = e[i] * inv;
  }
  __syncthreads();

  // feat[t,o] = sum_s v[s,o]*amap[s,t]; relu; write f16 concat tensor
  for (int o = 0; o < CQD; ++o) {
    float sum = 0.f;
    for (int s = 0; s < SDIM; ++s)
      sum += s_v[(s * 16 + x) * 32 + o] * s_att[(s * SDIM + t) * 16 + x];
    size_t gi = ((size_t)(t * NC + n * CQD + o) << 12) + (x << 8) + yz;
    sa[gi] = (_Float16)fmaxf(sum, 0.f);
  }
}

// ---------------------------------------------------------------------------
// Kernel 3/5: per-row mean / rsqrt(var+eps) over 4096 elements
// ---------------------------------------------------------------------------
__global__ void __launch_bounds__(256)
stat_kernel(const void* __restrict__ src, int is_half, float eps,
            float* __restrict__ mean, float* __restrict__ rstd) {
  __shared__ float s_sum[256], s_sq[256];
  size_t base = (size_t)blockIdx.x << 12;
  int tid = threadIdx.x;
  float sum = 0.f, sq = 0.f;
  for (int i = tid; i < 4096; i += 256) {
    float v = is_half ? (float)((const _Float16*)src)[base + i]
                      : ((const float*)src)[base + i];
    sum += v; sq += v * v;
  }
  s_sum[tid] = sum; s_sq[tid] = sq;
  __syncthreads();
  for (int st = 128; st > 0; st >>= 1) {
    if (tid < st) { s_sum[tid] += s_sum[tid + st]; s_sq[tid] += s_sq[tid + st]; }
    __syncthreads();
  }
  if (tid == 0) {
    float m  = s_sum[0] * (1.f / 4096.f);
    float vr = s_sq[0] * (1.f / 4096.f) - m * m;
    mean[blockIdx.x] = m;
    rstd[blockIdx.x] = rsqrtf(vr + eps);
  }
}

// ---------------------------------------------------------------------------
// Kernel 4: W1 GEMM [32 x 1024] with fused instance-norm on B, bias + relu
// one wave per 16x16 output tile, 32 WMMA k-steps
// ---------------------------------------------------------------------------
__global__ void __launch_bounds__(256)
w1_kernel(const _Float16* __restrict__ sa,
          const float* __restrict__ mean1, const float* __restrict__ rstd1,
          const _Float16* __restrict__ w116, const float* __restrict__ b1,
          float* __restrict__ out1) {
  int tid = threadIdx.x, lane = tid & 31, wid = tid >> 5;
  int tile = blockIdx.x * 8 + wid;          // 4096 tiles total
  int t  = tile >> 9;
  int ot = (tile >> 8) & 1;
  int pt = tile & 255;
  int o0 = ot * 16, p0 = pt * 16;
  int m     = lane & 15;
  int nn    = lane & 15;
  int kbase = (lane & 16) ? 16 : 0;
  int moff  = (lane & 16) ? 8 : 0;
  union { v8f v; float f[8]; } acc = {};
  for (int kb = 0; kb < 32; ++kb) {
    union { v16h v; _Float16 h[16]; } afr, bfr;
#pragma unroll
    for (int e = 0; e < 16; ++e) {
      afr.h[e] = w116[(o0 + m) * NC + kb * 32 + a_kmap(e, lane)];
      int mi = t * NC + kb * 32 + a_kmap(e, lane);
      float v = ((float)sa[((size_t)mi << 12) + p0 + nn] - mean1[mi]) * rstd1[mi];
      bfr.h[e] = (_Float16)v;
    }
    acc.v = __builtin_amdgcn_wmma_f32_16x16x32_f16(false, afr.v, false, bfr.v,
                                                   (short)0, acc.v, false, false);
  }
#pragma unroll
  for (int r = 0; r < 8; ++r) {
    int o = o0 + moff + r;
    out1[((size_t)(t * CQD + o) << 12) + p0 + nn] = fmaxf(acc.f[r] + b1[o], 0.f);
  }
}

// ---------------------------------------------------------------------------
// Kernel 6: W2 GEMM [32 x 32], K=32 -> single WMMA step, fused norm + bias
// ---------------------------------------------------------------------------
__global__ void __launch_bounds__(256)
w2_kernel(const float* __restrict__ h1,
          const float* __restrict__ mean2, const float* __restrict__ rstd2,
          const _Float16* __restrict__ w216, const float* __restrict__ b2,
          float* __restrict__ out) {
  int tid = threadIdx.x, lane = tid & 31, wid = tid >> 5;
  int tile = blockIdx.x * 8 + wid;
  int t  = tile >> 9;
  int ot = (tile >> 8) & 1;
  int pt = tile & 255;
  int o0 = ot * 16, p0 = pt * 16;
  int m     = lane & 15;
  int nn    = lane & 15;
  int kbase = (lane & 16) ? 16 : 0;
  int moff  = (lane & 16) ? 8 : 0;
  union { v16h v; _Float16 h[16]; } afr, bfr;
  union { v8f v; float f[8]; } acc = {};
#pragma unroll
  for (int e = 0; e < 16; ++e) {
    afr.h[e] = w216[(o0 + m) * CQD + a_kmap(e, lane)];
    int mi = t * CQD + a_kmap(e, lane);
    float v = (h1[((size_t)mi << 12) + p0 + nn] - mean2[mi]) * rstd2[mi];
    bfr.h[e] = (_Float16)v;
  }
  acc.v = __builtin_amdgcn_wmma_f32_16x16x32_f16(false, afr.v, false, bfr.v,
                                                 (short)0, acc.v, false, false);
#pragma unroll
  for (int r = 0; r < 8; ++r) {
    int o = o0 + moff + r;
    out[((size_t)(t * CQD + o) << 12) + p0 + nn] = acc.f[r] + b2[o];
  }
}

// ---------------------------------------------------------------------------
extern "C" void kernel_launch(void* const* d_in, const int* in_sizes, int n_in,
                              void* d_out, int out_size, void* d_ws, size_t ws_size,
                              hipStream_t stream) {
  const float* key   = (const float*)d_in[0];
  const float* query = (const float*)d_in[1];
  const float* gk_w  = (const float*)d_in[2];
  const float* gk_b  = (const float*)d_in[3];
  const float* gq_w  = (const float*)d_in[4];
  const float* gq_b  = (const float*)d_in[5];
  const float* Wk    = (const float*)d_in[6];
  const float* bk    = (const float*)d_in[7];
  const float* Wq    = (const float*)d_in[8];
  const float* bq    = (const float*)d_in[9];
  const float* Wv    = (const float*)d_in[10];
  const float* bv    = (const float*)d_in[11];
  const float* W1    = (const float*)d_in[12];
  const float* b1    = (const float*)d_in[13];
  const float* W2    = (const float*)d_in[14];
  const float* b2    = (const float*)d_in[15];
  float* out = (float*)d_out;

  // workspace layout
  _Float16* nk16  = (_Float16*)d_ws;                         // 1M halves (2 MiB)
  _Float16* nq16  = nk16 + (size_t)SDIM * CKD * SP;          // 1M halves
  _Float16* sa    = nq16 + (size_t)SDIM * CKD * SP;          // 32M halves (64 MiB)
  float*    mean1 = (float*)(sa + (size_t)SDIM * NC * SP);   // 8192
  float*    rstd1 = mean1 + SDIM * NC;                       // 8192
  float*    out1  = rstd1 + SDIM * NC;                       // 1M f32 (4 MiB)
  float*    mean2 = out1 + (size_t)SDIM * CQD * SP;          // 256
  float*    rstd2 = mean2 + SDIM * CQD;                      // 256
  _Float16* wk16  = (_Float16*)(rstd2 + SDIM * CQD);         // 16384
  _Float16* wq16  = wk16 + ND * AD * CKD;                    // 16384
  _Float16* wv16  = wq16 + ND * AD * CKD;                    // 32768
  _Float16* w116  = wv16 + ND * CQD * CKD;                   // 32768
  _Float16* w216  = w116 + CQD * NC;                         // 1024

  // 0. one-time weight conversion to f16
  cvt_kernel<<<64, 256, 0, stream>>>(Wk, wk16, ND * AD * CKD);
  cvt_kernel<<<64, 256, 0, stream>>>(Wq, wq16, ND * AD * CKD);
  cvt_kernel<<<128, 256, 0, stream>>>(Wv, wv16, ND * CQD * CKD);
  cvt_kernel<<<128, 256, 0, stream>>>(W1, w116, CQD * NC);
  cvt_kernel<<<4, 256, 0, stream>>>(W2, w216, CQD * CQD);
  // 1. group norms -> f16, transposed so each (y,z) tile is contiguous
  gn_kernel<<<16, 256, 0, stream>>>(key, query, gk_w, gk_b, gq_w, gq_b, nk16, nq16);
  // 2. fused per-(head,y,z) attention: async LDS fill + WMMA projections
  attn_kernel<<<ND * 256, 128, 0, stream>>>(nk16, nq16, wk16, bk, wq16, bq, wv16, bv, sa);
  // 3. instance-norm stats over concat tensor (eps = N*CQ = 1024 per reference)
  stat_kernel<<<SDIM * NC, 256, 0, stream>>>(sa, 1, 1024.0f, mean1, rstd1);
  // 4. W1 GEMM + bias + relu (norm fused into B-fragment load)
  w1_kernel<<<512, 256, 0, stream>>>(sa, mean1, rstd1, w116, b1, out1);
  // 5. second instance-norm stats
  stat_kernel<<<SDIM * CQD, 256, 0, stream>>>(out1, 0, 1e-5f, mean2, rstd2);
  // 6. W2 GEMM + bias -> final output [1,8,32,16,16,16]
  w2_kernel<<<512, 256, 0, stream>>>(out1, mean2, rstd2, w216, b2, out);
}
